// SelectiveQuantizedMLP_47184510714571
// MI455X (gfx1250) — 8-bit matrix operand order repaired and checked by reading, compiled, not run on hardware
//
#include <hip/hip_runtime.h>
#include <hip/hip_bf16.h>
#include <math.h>

typedef __attribute__((ext_vector_type(8))) int v8i;

__device__ __forceinline__ unsigned encf(float f) {
  unsigned u = __float_as_uint(f);
  return (u & 0x80000000u) ? ~u : (u | 0x80000000u);
}
__device__ __forceinline__ float decf(unsigned e) {
  unsigned u = (e & 0x80000000u) ? (e ^ 0x80000000u) : ~e;
  return __uint_as_float(u);
}

__global__ void k_init(unsigned* s) {
  s[0] = 0xFFFFFFFFu; s[1] = 0u; s[2] = 0u; s[3] = 0u; s[4] = 0xFFFFFFFFu; s[5] = 0u;
}

__global__ __launch_bounds__(256) void k_minmax(const float4* __restrict__ p, long n4,
                                                unsigned* emin, unsigned* emax) {
  const int tid = threadIdx.x;
  float lmin = INFINITY, lmax = -INFINITY;
  for (long i = (long)blockIdx.x * blockDim.x + tid; i < n4; i += (long)gridDim.x * blockDim.x) {
    float4 v = p[i];
    lmin = fminf(lmin, fminf(fminf(v.x, v.y), fminf(v.z, v.w)));
    lmax = fmaxf(lmax, fmaxf(fmaxf(v.x, v.y), fmaxf(v.z, v.w)));
  }
  __shared__ float smn[256], smx[256];
  smn[tid] = lmin; smx[tid] = lmax;
  __syncthreads();
  for (int s = 128; s > 0; s >>= 1) {
    if (tid < s) {
      smn[tid] = fminf(smn[tid], smn[tid + s]);
      smx[tid] = fmaxf(smx[tid], smx[tid + s]);
    }
    __syncthreads();
  }
  if (tid == 0) {
    atomicMin(emin, encf(smn[0]));
    atomicMax(emax, encf(smx[0]));
  }
}

__global__ __launch_bounds__(256) void k_absmax(const float4* __restrict__ p, long n4,
                                                unsigned* emax) {
  const int tid = threadIdx.x;
  float lmax = 0.0f;
  for (long i = (long)blockIdx.x * blockDim.x + tid; i < n4; i += (long)gridDim.x * blockDim.x) {
    float4 v = p[i];
    lmax = fmaxf(lmax, fmaxf(fmaxf(fabsf(v.x), fabsf(v.y)), fmaxf(fabsf(v.z), fabsf(v.w))));
  }
  __shared__ float smx[256];
  smx[tid] = lmax;
  __syncthreads();
  for (int s = 128; s > 0; s >>= 1) {
    if (tid < s) smx[tid] = fmaxf(smx[tid], smx[tid + s]);
    __syncthreads();
  }
  if (tid == 0) atomicMax(emax, encf(smx[0]));
}

__global__ void k_scales1(const unsigned* s, float* f) {
  float xmin = decf(s[0]), xmax = decf(s[1]);
  float sx = fmaxf((xmax - xmin) / 255.0f, 1e-8f);
  float zx = fminf(fmaxf(rintf(-xmin / sx) - 128.0f, -128.0f), 127.0f);
  f[8] = sx; f[9] = zx;
  f[10] = fmaxf(decf(s[2]) / 127.0f, 1e-8f);
  f[11] = fmaxf(decf(s[3]) / 127.0f, 1e-8f);
}
__global__ void k_scales_h(const unsigned* s, float* f) {
  float hmin = decf(s[4]), hmax = decf(s[5]);
  float sh = fmaxf((hmax - hmin) / 255.0f, 1e-8f);
  float zh = fminf(fmaxf(rintf(-hmin / sh) - 128.0f, -128.0f), 127.0f);
  f[12] = sh; f[13] = zh;
}

__global__ __launch_bounds__(256) void k_quant_act(const float4* __restrict__ in,
                                                   int* __restrict__ outq, long n4,
                                                   const float* __restrict__ scalf,
                                                   int si, int zi) {
  const float inv = 1.0f / scalf[si];
  const float zp  = scalf[zi];
  for (long i = (long)blockIdx.x * blockDim.x + threadIdx.x; i < n4;
       i += (long)gridDim.x * blockDim.x) {
    float4 v = in[i];
    int q0 = (int)fminf(fmaxf(rintf(v.x * inv) + zp, -128.0f), 127.0f);
    int q1 = (int)fminf(fmaxf(rintf(v.y * inv) + zp, -128.0f), 127.0f);
    int q2 = (int)fminf(fmaxf(rintf(v.z * inv) + zp, -128.0f), 127.0f);
    int q3 = (int)fminf(fmaxf(rintf(v.w * inv) + zp, -128.0f), 127.0f);
    outq[i] = (q0 & 0xFF) | ((q1 & 0xFF) << 8) | ((q2 & 0xFF) << 16) | (q3 << 24);
  }
}

__global__ __launch_bounds__(256) void k_quant_wrow(const float* __restrict__ W,
                                                    signed char* __restrict__ wq,
                                                    int* __restrict__ wsum, int K,
                                                    const float* __restrict__ scalf, int si) {
  const int row = blockIdx.x;
  const int tid = threadIdx.x;
  const float inv = 1.0f / scalf[si];
  const float* wr = W + (size_t)row * K;
  signed char* qr = wq + (size_t)row * K;
  int sum = 0;
  for (int k = tid; k < K; k += blockDim.x) {
    int q = (int)fminf(fmaxf(rintf(wr[k] * inv), -127.0f), 127.0f);
    qr[k] = (signed char)q;
    sum += q;
  }
  __shared__ int ssum[256];
  ssum[tid] = sum;
  __syncthreads();
  for (int s = 128; s > 0; s >>= 1) {
    if (tid < s) ssum[tid] += ssum[tid + s];
    __syncthreads();
  }
  if (tid == 0) wsum[row] = ssum[0];
}

__device__ __forceinline__ v8i load_a_frag(const signed char* p) {
  const int2 d0 = *(const int2*)(p + 0);
  const int2 d1 = *(const int2*)(p + 16);
  const int2 d2 = *(const int2*)(p + 32);
  const int2 d3 = *(const int2*)(p + 48);
  v8i r;
  r[0] = d0.x; r[1] = d0.y; r[2] = d1.x; r[3] = d1.y;
  r[4] = d2.x; r[5] = d2.y; r[6] = d3.x; r[7] = d3.y;
  return r;
}
__device__ __forceinline__ v8i load_b_frag(const signed char* p) {
  const int2 d0 = *(const int2*)(p + 0);
  const int2 d1 = *(const int2*)(p + 16);
  const int2 d2 = *(const int2*)(p + 32);
  const int2 d3 = *(const int2*)(p + 48);
  v8i r;
  r[0] = d0.x; r[1] = d0.y; r[2] = d1.x; r[3] = d1.y;
  r[4] = d2.x; r[5] = d2.y; r[6] = d3.x; r[7] = d3.y;
  return r;
}

template <int DO_GELU>
__global__ __launch_bounds__(256) void k_qgemm(
    const signed char* __restrict__ Aq, const signed char* __restrict__ Wq,
    const int* __restrict__ wsum, const float* __restrict__ bias,
    const float* __restrict__ scalf, int i_sa, int i_sw, int i_zp,
    float* __restrict__ out, int M, int N, int K,
    unsigned* emin, unsigned* emax) {
  const int tid  = threadIdx.x;
  const int lane = tid & 31;
  const int wave = tid >> 5;
  const int wm = wave >> 1, wn = wave & 1;
  const int half = lane >> 4;
  const int lidx = lane & 15;

  const int m0 = blockIdx.y * 128 + wm * 32;
  const int n0 = blockIdx.x * 128 + wn * 64;

  v8i creg[8];
#pragma unroll
  for (int i = 0; i < 8; ++i) creg[i] = (v8i){0, 0, 0, 0, 0, 0, 0, 0};

  const signed char* arow0 = Aq + (size_t)(m0 + lidx) * K + half * 8;
  const signed char* arow1 = arow0 + (size_t)16 * K;
  const signed char* brow  = Wq + (size_t)(n0 + lidx) * K + half * 8;

  for (int k0 = 0; k0 < K; k0 += 64) {
    if (k0 + 64 < K) {
      __builtin_prefetch(arow0 + k0 + 64, 0, 3);
      __builtin_prefetch(brow + k0 + 64, 0, 3);
    }
    v8i a0 = load_a_frag(arow0 + k0);
    v8i a1 = load_a_frag(arow1 + k0);
    v8i bf[4];
#pragma unroll
    for (int fj = 0; fj < 4; ++fj)
      bf[fj] = load_b_frag(brow + (size_t)(fj * 16) * K + k0);
#pragma unroll
    for (int fj = 0; fj < 4; ++fj) {
      creg[fj]     = __builtin_amdgcn_wmma_i32_16x16x64_iu8(true, a0, true, bf[fj], creg[fj],     false, false);
      creg[4 + fj] = __builtin_amdgcn_wmma_i32_16x16x64_iu8(true, a1, true, bf[fj], creg[4 + fj], false, false);
    }
  }

  const float dq = scalf[i_sa] * scalf[i_sw];
  const int zp = (int)scalf[i_zp];
  float lmin = INFINITY, lmax = -INFINITY;

#pragma unroll
  for (int fi = 0; fi < 2; ++fi) {
#pragma unroll
    for (int fj = 0; fj < 4; ++fj) {
      v8i c = creg[fi * 4 + fj];
      const int col = n0 + fj * 16 + lidx;
      const int wsu = wsum[col];
      const float bb = bias[col];
      const int rowb = m0 + fi * 16 + 8 * half;
#pragma unroll
      for (int r = 0; r < 8; ++r) {
        float v = (float)(c[r] - zp * wsu) * dq + bb;
        if (DO_GELU) {
          v = 0.5f * v * (1.0f + erff(v * 0.70710678118654752f));
          lmin = fminf(lmin, v);
          lmax = fmaxf(lmax, v);
        }
        out[(size_t)(rowb + r) * N + col] = v;
      }
    }
  }

  if (DO_GELU) {
    __shared__ float smn[256], smx[256];
    smn[tid] = lmin; smx[tid] = lmax;
    __syncthreads();
    for (int s = 128; s > 0; s >>= 1) {
      if (tid < s) {
        smn[tid] = fminf(smn[tid], smn[tid + s]);
        smx[tid] = fmaxf(smx[tid], smx[tid + s]);
      }
      __syncthreads();
    }
    if (tid == 0) {
      atomicMin(emin, encf(smn[0]));
      atomicMax(emax, encf(smx[0]));
    }
  }
}

extern "C" void kernel_launch(void* const* d_in, const int* in_sizes, int n_in,
                              void* d_out, int out_size, void* d_ws, size_t ws_size,
                              hipStream_t stream) {
  const float* x  = (const float*)d_in[0];
  const float* W1 = (const float*)d_in[1];
  const float* b1 = (const float*)d_in[2];
  const float* W2 = (const float*)d_in[3];
  const float* b2 = (const float*)d_in[4];

  const int d = in_sizes[4];
  const int h = in_sizes[2];
  const long Mtok = (long)in_sizes[0] / d;

  unsigned char* ws = (unsigned char*)d_ws;
  unsigned* scal_u = (unsigned*)ws;
  float*    scal_f = (float*)ws;
  int* wsum1 = (int*)(ws + 1024);
  int* wsum2 = (int*)(ws + 1024 + (size_t)h * 4);
  signed char* xq  = (signed char*)(ws + 65536);
  signed char* wq1 = xq + (size_t)Mtok * d;
  signed char* wq2 = wq1 + (size_t)h * d;
  signed char* hq  = wq2 + (size_t)d * h;
  float* hbuf = (float*)(hq + (size_t)Mtok * h);

  k_init<<<1, 1, 0, stream>>>(scal_u);

  const long nx4 = (long)Mtok * d / 4;
  k_minmax<<<1024, 256, 0, stream>>>((const float4*)x, nx4, &scal_u[0], &scal_u[1]);
  k_absmax<<<512, 256, 0, stream>>>((const float4*)W1, (long)h * d / 4, &scal_u[2]);
  k_absmax<<<512, 256, 0, stream>>>((const float4*)W2, (long)d * h / 4, &scal_u[3]);
  k_scales1<<<1, 1, 0, stream>>>(scal_u, scal_f);

  k_quant_act<<<2048, 256, 0, stream>>>((const float4*)x, (int*)xq, nx4, scal_f, 8, 9);
  k_quant_wrow<<<h, 256, 0, stream>>>(W1, wq1, wsum1, d, scal_f, 10);
  k_quant_wrow<<<d, 256, 0, stream>>>(W2, wq2, wsum2, h, scal_f, 11);

  dim3 g1(h / 128, (unsigned)(Mtok / 128));
  k_qgemm<1><<<g1, 256, 0, stream>>>(xq, wq1, wsum1, b1, scal_f, 8, 10, 9,
                                     hbuf, (int)Mtok, h, d, &scal_u[4], &scal_u[5]);

  k_scales_h<<<1, 1, 0, stream>>>(scal_u, scal_f);
  k_quant_act<<<2048, 256, 0, stream>>>((const float4*)hbuf, (int*)hq,
                                        (long)Mtok * h / 4, scal_f, 12, 13);

  dim3 g2(d / 128, (unsigned)(Mtok / 128));
  k_qgemm<0><<<g2, 256, 0, stream>>>(hq, wq2, wsum2, b2, scal_f, 12, 11, 13,
                                     (float*)d_out, (int)Mtok, d, h, nullptr, nullptr);
}
